// BatchedMoE_86904368268077
// MI455X (gfx1250) — matrix operand order repaired and checked by reading, not compiled, not run on hardware
//
#include <hip/hip_runtime.h>

// ---------------------------------------------------------------------------
// MoE (top-2 of 8 experts, SwiGLU) + shared expert, fp32 in/out.
// Compute path: bf16 WMMA (v_wmma_f32_16x16x32_bf16) with f32 accumulation,
// double-buffered LDS pipeline with register staging.
// N=2048 tokens, C=2048, E=8, I=1408, S=2816.
// ---------------------------------------------------------------------------

typedef __attribute__((ext_vector_type(16))) __bf16 v16bf;
typedef __attribute__((ext_vector_type(2)))  __bf16 v2bf;
typedef __attribute__((ext_vector_type(8)))  float  v8f;

#define BM 64
#define BN 128
#define BK 32
#define NTHREADS 256

// ----- f32 -> bf16x2 pack (compiler should select v_cvt_pk_bf16_f32) -------
__device__ __forceinline__ unsigned pack2(float a, float b) {
    v2bf t;
    t.x = (__bf16)a;
    t.y = (__bf16)b;
    return __builtin_bit_cast(unsigned, t);
}

__device__ __forceinline__ float fast_rcp(float x) {
#if __has_builtin(__builtin_amdgcn_rcpf)
    return __builtin_amdgcn_rcpf(x);
#else
    return 1.f / x;
#endif
}

// ----- WMMA fragment loaders from LDS (per CDNA5 ISA 7.12.2 layouts) -------
// A: 16x32 bf16. lane l: m=l&15, h=l>>4.
//    v<4 : K = 8h+2v       -> dword idx 4h+v
//    v>=4: K = 16+8h+2(v-4)-> dword idx 8+4h+(v-4)
__device__ __forceinline__ v16bf load_a_frag(const __bf16* As, int row0, int lane) {
    int m = lane & 15, h = lane >> 4;
    const unsigned* p = (const unsigned*)(As + (row0 + m) * BK);
    v16bf a;
    unsigned* ap = (unsigned*)&a;
#pragma unroll
    for (int v = 0; v < 4; ++v) ap[v] = p[4 * h + v];
#pragma unroll
    for (int v = 0; v < 4; ++v) ap[4 + v] = p[8 + 4 * h + v];
    return a;
}
// B: 32x16 bf16. lane l: n=l&15, h=l>>4; same K order as A: v<4 -> dword 4h+v, v>=4 -> dword 8+4h+(v-4).
__device__ __forceinline__ v16bf load_b_frag(const __bf16* Bs, int col0, int lane) {
    int n = lane & 15, h = lane >> 4;
    const unsigned* p = (const unsigned*)(Bs + (col0 + n) * BK);
    v16bf b;
    unsigned* bp = (unsigned*)&b;
#pragma unroll
    for (int v = 0; v < 4; ++v) { bp[v] = p[4 * h + v]; bp[4 + v] = p[8 + 4 * h + v]; }
    return b;
}

#define WMMA_BF16(A, B, Cacc) \
    __builtin_amdgcn_wmma_f32_16x16x32_bf16(false, (A), false, (B), (short)0, (Cacc), false, false)

// ---------------------------------------------------------------------------
// Routing kernels
// ---------------------------------------------------------------------------
__global__ void moe_init_kernel(int* cnt) {
    if (threadIdx.x < 8) cnt[threadIdx.x] = 0;
}

__global__ void __launch_bounds__(256)
moe_gate_kernel(const float* __restrict__ X, const float* __restrict__ GW,
                int* __restrict__ cnt, int* __restrict__ petok,
                float* __restrict__ peprob, int N, int C) {
    int wave = threadIdx.x >> 5, lane = threadIdx.x & 31;
    int tok = blockIdx.x * 8 + wave;
    if (tok >= N) return;
    float s[8] = {0.f, 0.f, 0.f, 0.f, 0.f, 0.f, 0.f, 0.f};
    const float* xr = X + (size_t)tok * C;
    for (int c = lane; c < C; c += 32) {
        float xv = xr[c];
#pragma unroll
        for (int e = 0; e < 8; ++e) s[e] += xv * GW[e * C + c];
    }
#pragma unroll
    for (int off = 16; off > 0; off >>= 1) {
#pragma unroll
        for (int e = 0; e < 8; ++e) s[e] += __shfl_down(s[e], off, 32);
    }
    if (lane == 0) {
        int i0 = 0; float v0 = s[0];
#pragma unroll
        for (int e = 1; e < 8; ++e) if (s[e] > v0) { v0 = s[e]; i0 = e; }
        int i1 = (i0 == 0) ? 1 : 0; float v1 = s[i1];
#pragma unroll
        for (int e = 0; e < 8; ++e)
            if (e != i0 && s[e] > v1) { v1 = s[e]; i1 = e; }
        // softmax over the two selected logits (v0 >= v1)
        float e1 = __expf(v1 - v0);
        float p0 = fast_rcp(1.f + e1);
        float p1 = e1 * p0;
        int pos0 = atomicAdd(&cnt[i0], 1);
        petok[i0 * N + pos0] = tok;
        peprob[i0 * N + pos0] = p0;
        int pos1 = atomicAdd(&cnt[i1], 1);
        petok[i1 * N + pos1] = tok;
        peprob[i1 * N + pos1] = p1;
    }
}

__global__ void __launch_bounds__(256)
moe_compact_kernel(const int* __restrict__ cnt, int* __restrict__ offs,
                   const int* __restrict__ petok, const float* __restrict__ peprob,
                   int* __restrict__ rows_flat, float* __restrict__ prob_flat, int N) {
    __shared__ int so[9];
    if (threadIdx.x == 0) {
        int acc = 0;
        for (int e = 0; e < 8; ++e) { so[e] = acc; offs[e] = acc; acc += cnt[e]; }
        so[8] = acc; offs[8] = acc;
    }
    __syncthreads();
    for (int e = 0; e < 8; ++e) {
        int c = cnt[e], b = so[e];
        for (int i = threadIdx.x; i < c; i += blockDim.x) {
            rows_flat[b + i] = petok[e * N + i];
            prob_flat[b + i] = peprob[e * N + i];
        }
    }
}

// ---------------------------------------------------------------------------
// Stage 1: H = silu(A * W1^T) * (A * W3^T)   (A rows optionally gathered)
//   A: X [Ntok][K=C], W1/W3: [(E)][Iw][K] row-major, H: [rowsTotal][Iw]
// Double-buffered LDS pipeline: global loads for tile k+1 issue before the
// WMMA stream for tile k, hiding HBM/L2 latency behind the matrix pipe.
// ---------------------------------------------------------------------------
template <bool GATHER>
__global__ void __launch_bounds__(NTHREADS)
moe_hswiglu_kernel(const float* __restrict__ X, const float* __restrict__ W1,
                   const float* __restrict__ W3, float* __restrict__ H,
                   const int* __restrict__ rows_flat, const int* __restrict__ cnts,
                   const int* __restrict__ offs, int K, int Iw, int fixedM) {
    const int e = blockIdx.z;
    int M, rowBase;
    if (GATHER) { M = cnts[e]; rowBase = offs[e]; }
    else        { M = fixedM;  rowBase = 0; }
    const int tileM = blockIdx.y * BM;
    if (tileM >= M) return;
    const int tileN = blockIdx.x * BN;
    const float* B1 = W1 + (size_t)e * Iw * K;
    const float* B3 = W3 + (size_t)e * Iw * K;

    __shared__ __bf16 As[2][BM * BK];
    __shared__ __bf16 B1s[2][BN * BK];
    __shared__ __bf16 B3s[2][BN * BK];

    const int tid = threadIdx.x;
    const int lane = tid & 31;
    const int w = tid >> 5;
    const int wm = w >> 2, wn = w & 3;   // 2x4 waves; each owns 32x32

    v8f acc1[2][2] = {};
    v8f acc3[2][2] = {};

    // per-thread staging coordinates (fixed across k-steps)
    const int ar = tid >> 2;                 // 0..63
    const int ak = (tid & 3) * 8;            // 0,8,16,24
    int am = tileM + ar;
    int amc = (am < M) ? am : (M - 1);
    int tok;
    if (GATHER) tok = rows_flat[rowBase + amc];
    else        tok = amc;
    const float* aRow = X + (size_t)tok * K + ak;

    const int bn = tid >> 1;                 // 0..127
    const int bk = (tid & 1) * 16;           // 0,16
    const float* b1Row = B1 + (size_t)(tileN + bn) * K + bk;
    const float* b3Row = B3 + (size_t)(tileN + bn) * K + bk;

    float4 ra[2], rb1[4], rb3[4];
    auto load_regs = [&](int k0) {
        const float4* sa = (const float4*)(aRow + k0);
        ra[0] = sa[0]; ra[1] = sa[1];
        const float4* s1 = (const float4*)(b1Row + k0);
        const float4* s3 = (const float4*)(b3Row + k0);
#pragma unroll
        for (int i = 0; i < 4; ++i) { rb1[i] = s1[i]; rb3[i] = s3[i]; }
    };
    auto store_lds = [&](int buf) {
        unsigned* da = (unsigned*)(&As[buf][ar * BK + ak]);
#pragma unroll
        for (int i = 0; i < 2; ++i) {
            da[2 * i]     = pack2(ra[i].x, ra[i].y);
            da[2 * i + 1] = pack2(ra[i].z, ra[i].w);
        }
        unsigned* d1 = (unsigned*)(&B1s[buf][bn * BK + bk]);
        unsigned* d3 = (unsigned*)(&B3s[buf][bn * BK + bk]);
#pragma unroll
        for (int i = 0; i < 4; ++i) {
            d1[2 * i]     = pack2(rb1[i].x, rb1[i].y);
            d1[2 * i + 1] = pack2(rb1[i].z, rb1[i].w);
            d3[2 * i]     = pack2(rb3[i].x, rb3[i].y);
            d3[2 * i + 1] = pack2(rb3[i].z, rb3[i].w);
        }
    };

    const int nk = K / BK;
    load_regs(0);
    store_lds(0);
    __syncthreads();

    for (int kt = 0; kt < nk; ++kt) {
        const int cur = kt & 1;
        const bool more = (kt + 1 < nk);
        if (more) load_regs((kt + 1) * BK);   // overlap with WMMA below

        v16bf af0 = load_a_frag(As[cur], wm * 32, lane);
        v16bf af1 = load_a_frag(As[cur], wm * 32 + 16, lane);
#pragma unroll
        for (int j = 0; j < 2; ++j) {
            v16bf bf1 = load_b_frag(B1s[cur], wn * 32 + j * 16, lane);
            v16bf bf3 = load_b_frag(B3s[cur], wn * 32 + j * 16, lane);
            acc1[0][j] = WMMA_BF16(af0, bf1, acc1[0][j]);
            acc1[1][j] = WMMA_BF16(af1, bf1, acc1[1][j]);
            acc3[0][j] = WMMA_BF16(af0, bf3, acc3[0][j]);
            acc3[1][j] = WMMA_BF16(af1, bf3, acc3[1][j]);
        }
        if (more) {
            __syncthreads();
            store_lds(cur ^ 1);
            __syncthreads();
        }
    }

    // epilogue: h = silu(acc1) * acc3  (fast sigmoid via v_rcp_f32)
    const int nCol = lane & 15, hh = lane >> 4;
#pragma unroll
    for (int i = 0; i < 2; ++i) {
#pragma unroll
        for (int j = 0; j < 2; ++j) {
#pragma unroll
            for (int r = 0; r < 8; ++r) {
                int m = tileM + wm * 32 + i * 16 + r + 8 * hh;
                if (m < M) {
                    int col = tileN + wn * 32 + j * 16 + nCol;
                    float a = acc1[i][j][r];
                    float b = acc3[i][j][r];
                    float hv = a * fast_rcp(1.f + __expf(-a)) * b;
                    H[(size_t)(rowBase + m) * Iw + col] = hv;
                }
            }
        }
    }
}

// ---------------------------------------------------------------------------
// Stage 2: O = A * W^T ; plain store (shared expert) or prob-scaled atomic
// scatter-add onto Y[token] (routed experts).
//   A: [rowsTotal][K], W: [(E)][Cout][K] row-major, Y: [Ntok][Cout]
// ---------------------------------------------------------------------------
template <bool SCATTER>
__global__ void __launch_bounds__(NTHREADS)
moe_outgemm_kernel(const float* __restrict__ A, const float* __restrict__ W,
                   float* __restrict__ Y, const int* __restrict__ rows_flat,
                   const float* __restrict__ prob_flat, const int* __restrict__ cnts,
                   const int* __restrict__ offs, int K, int Cout, int fixedM) {
    const int e = blockIdx.z;
    int M, rowBase;
    if (SCATTER) { M = cnts[e]; rowBase = offs[e]; }
    else         { M = fixedM;  rowBase = 0; }
    const int tileM = blockIdx.y * BM;
    if (tileM >= M) return;
    const int tileN = blockIdx.x * BN;
    const float* B = W + (size_t)e * Cout * K;

    __shared__ __bf16 As[2][BM * BK];
    __shared__ __bf16 Bs[2][BN * BK];

    const int tid = threadIdx.x;
    const int lane = tid & 31;
    const int w = tid >> 5;
    const int wm = w >> 2, wn = w & 3;

    v8f acc[2][2] = {};

    const int ar = tid >> 2;
    const int ak = (tid & 3) * 8;
    int am = tileM + ar;
    int amc = (am < M) ? am : (M - 1);
    const float* aRow = A + (size_t)(rowBase + amc) * K + ak;

    const int bn = tid >> 1;
    const int bk = (tid & 1) * 16;
    const float* bRow = B + (size_t)(tileN + bn) * K + bk;

    float4 ra[2], rb[4];
    auto load_regs = [&](int k0) {
        const float4* sa = (const float4*)(aRow + k0);
        ra[0] = sa[0]; ra[1] = sa[1];
        const float4* sb = (const float4*)(bRow + k0);
#pragma unroll
        for (int i = 0; i < 4; ++i) rb[i] = sb[i];
    };
    auto store_lds = [&](int buf) {
        unsigned* da = (unsigned*)(&As[buf][ar * BK + ak]);
#pragma unroll
        for (int i = 0; i < 2; ++i) {
            da[2 * i]     = pack2(ra[i].x, ra[i].y);
            da[2 * i + 1] = pack2(ra[i].z, ra[i].w);
        }
        unsigned* db = (unsigned*)(&Bs[buf][bn * BK + bk]);
#pragma unroll
        for (int i = 0; i < 4; ++i) {
            db[2 * i]     = pack2(rb[i].x, rb[i].y);
            db[2 * i + 1] = pack2(rb[i].z, rb[i].w);
        }
    };

    const int nk = K / BK;
    load_regs(0);
    store_lds(0);
    __syncthreads();

    for (int kt = 0; kt < nk; ++kt) {
        const int cur = kt & 1;
        const bool more = (kt + 1 < nk);
        if (more) load_regs((kt + 1) * BK);

        v16bf af0 = load_a_frag(As[cur], wm * 32, lane);
        v16bf af1 = load_a_frag(As[cur], wm * 32 + 16, lane);
#pragma unroll
        for (int j = 0; j < 2; ++j) {
            v16bf bf = load_b_frag(Bs[cur], wn * 32 + j * 16, lane);
            acc[0][j] = WMMA_BF16(af0, bf, acc[0][j]);
            acc[1][j] = WMMA_BF16(af1, bf, acc[1][j]);
        }
        if (more) {
            __syncthreads();
            store_lds(cur ^ 1);
            __syncthreads();
        }
    }

    const int nCol = lane & 15, hh = lane >> 4;
#pragma unroll
    for (int i = 0; i < 2; ++i) {
#pragma unroll
        for (int j = 0; j < 2; ++j) {
#pragma unroll
            for (int r = 0; r < 8; ++r) {
                int m = tileM + wm * 32 + i * 16 + r + 8 * hh;
                if (m < M) {
                    int col = tileN + wn * 32 + j * 16 + nCol;
                    float val = acc[i][j][r];
                    if (SCATTER) {
                        int s = rowBase + m;
                        int tk = rows_flat[s];
                        float p = prob_flat[s];
                        atomicAdd(&Y[(size_t)tk * Cout + col], val * p);
                    } else {
                        Y[(size_t)m * Cout + col] = val;
                    }
                }
            }
        }
    }
}

// ---------------------------------------------------------------------------
extern "C" void kernel_launch(void* const* d_in, const int* in_sizes, int n_in,
                              void* d_out, int out_size, void* d_ws, size_t ws_size,
                              hipStream_t stream) {
    const float* x      = (const float*)d_in[0];
    const float* gate_w = (const float*)d_in[1];
    const float* w1     = (const float*)d_in[2];
    const float* w3     = (const float*)d_in[3];
    const float* w2     = (const float*)d_in[4];
    const float* sw1    = (const float*)d_in[5];
    const float* sw3    = (const float*)d_in[6];
    const float* sw2    = (const float*)d_in[7];
    float* y = (float*)d_out;

    const int N = 2048, C = 2048, E = 8, I = 1408, S = 2816;
    (void)in_sizes; (void)n_in; (void)out_size; (void)ws_size;

    // workspace layout (~23.3 MB): counters, per-expert lists, flat compacted
    // routing, and one H buffer reused by shared and expert phases
    // (N*S == 2*N*I == 5,767,168 floats).
    int*   cnt       = (int*)d_ws;               // 8
    int*   offs      = cnt + 8;                  // 9 (padded to 16)
    int*   petok     = offs + 16;                // E*N
    float* peprob    = (float*)(petok + E * N);  // E*N
    int*   rows_flat = (int*)(peprob + E * N);   // 2*N
    float* prob_flat = (float*)(rows_flat + 2 * N); // 2*N
    float* Hbuf      = (float*)(prob_flat + 2 * N); // N*S floats

    // 1) routing
    moe_init_kernel<<<1, 32, 0, stream>>>(cnt);
    moe_gate_kernel<<<N / 8, 256, 0, stream>>>(x, gate_w, cnt, petok, peprob, N, C);
    moe_compact_kernel<<<1, 256, 0, stream>>>(cnt, offs, petok, peprob,
                                              rows_flat, prob_flat, N);

    // 2) shared expert: hs = silu(x sw1^T) * (x sw3^T); y = hs sw2^T (store)
    moe_hswiglu_kernel<false><<<dim3(S / BN, N / BM, 1), NTHREADS, 0, stream>>>(
        x, sw1, sw3, Hbuf, nullptr, nullptr, nullptr, C, S, N);
    moe_outgemm_kernel<false><<<dim3(C / BN, N / BM, 1), NTHREADS, 0, stream>>>(
        Hbuf, sw2, y, nullptr, nullptr, nullptr, nullptr, S, C, N);

    // 3) routed experts (grouped GEMM, tiles early-exit past per-expert count)
    moe_hswiglu_kernel<true><<<dim3(I / BN, N / BM, E), NTHREADS, 0, stream>>>(
        x, w1, w3, Hbuf, rows_flat, cnt, offs, C, I, 0);
    moe_outgemm_kernel<true><<<dim3(C / BN, N / BM, E), NTHREADS, 0, stream>>>(
        Hbuf, w2, y, rows_flat, prob_flat, cnt, offs, I, C, 0);
}
